// WhileOpLSTMNet_1984274890954
// MI455X (gfx1250) — matrix operand order repaired and checked by reading, not compiled, not run on hardware
//
#include <hip/hip_runtime.h>
#include <hip/hip_bf16.h>
#include <stdint.h>

// Problem dims (fixed by reference)
#define B_ 64
#define T_ 256
#define D_ 1024
#define L_ 4
#define H_ 1024
static_assert(D_ == H_, "code assumes D == H (uniform K stride 1024)");

typedef __attribute__((ext_vector_type(16))) __bf16 bf16x16;
typedef __attribute__((ext_vector_type(8)))  float  floatx8;

__device__ __forceinline__ uint16_t f32_to_bf16_rne(float f) {
  uint32_t u = __float_as_uint(f);
  uint32_t rnd = 0x7FFFu + ((u >> 16) & 1u);
  return (uint16_t)((u + rnd) >> 16);
}
__device__ __forceinline__ float bf16_to_f32(uint16_t v) {
  return __uint_as_float(((uint32_t)v) << 16);
}
__device__ __forceinline__ float fast_sigmoid(float x) {
  return 1.0f / (1.0f + __expf(-x));
}
__device__ __forceinline__ float fast_tanh(float x) {
  return 2.0f / (1.0f + __expf(-2.0f * x)) - 1.0f;
}

// WMMA operand fragment: 16 bf16 per lane (32 bytes) = two b128 loads
union Frag { uint4 q[2]; bf16x16 v; };

// A-operand (16-bit 16x32 layout): row = lane&15; K-half selected by lane>=16.
// arow must already include (+ row*stride + kbase + 8*kh).
__device__ __forceinline__ Frag ldA(const uint16_t* arow, int k) {
  Frag f;
  f.q[0] = *(const uint4*)(arow + k);        // K = k .. k+7   (+8*kh folded in)
  f.q[1] = *(const uint4*)(arow + k + 16);   // K = k+16 .. k+23
  return f;
}
// B-operand (16-bit 32x16 layout): col = lane&15; K-half by lane>=16.
// brow must already include (+ col*Kdim + kbase + 8*kh).
__device__ __forceinline__ Frag ldB(const uint16_t* brow, int k) {
  Frag f;
  f.q[0] = *(const uint4*)(brow + k);
  f.q[1] = *(const uint4*)(brow + k + 16);
  return f;
}

// ---------------- prep kernels ----------------

__global__ void cvt_f32_to_bf16(const float* __restrict__ in,
                                uint16_t* __restrict__ out, long n) {
  long i = (long)blockIdx.x * blockDim.x + threadIdx.x;
  if (i < n) out[i] = f32_to_bf16_rne(in[i]);
}

// in: [L][K][N] f32  ->  out: [L][N][K] bf16   (N = 4H)
__global__ void cvt_transpose_bf16(const float* __restrict__ in,
                                   uint16_t* __restrict__ out,
                                   int K, int N) {
  long i = (long)blockIdx.x * blockDim.x + threadIdx.x;
  long total = (long)L_ * K * N;
  if (i >= total) return;
  int n = (int)(i % N);
  long rem = i / N;
  int k = (int)(rem % K);
  int l = (int)(rem / K);
  out[((long)l * N + n) * K + k] = f32_to_bf16_rne(in[i]);
}

__global__ void zero_u32(uint32_t* __restrict__ p, long n) {
  long i = (long)blockIdx.x * blockDim.x + threadIdx.x;
  if (i < n) p[i] = 0u;
}

// ---------------- off-critical-path GEMM: gx0 = x @ Wx[0] ----------------
// A: xbf viewed as [B*T][D] row-major (row j = b*T + t). B: WxT0 [4H][D] bf16.
// Out: gx0 [T][B][4H] bf16. One wave = 16 rows x 64 cols (4 n-tiles share A).

__global__ __launch_bounds__(256)
void gemm_x_wx0(const uint16_t* __restrict__ xbf,
                const uint16_t* __restrict__ WxT0,
                uint16_t* __restrict__ gx0) {
  const int lane = threadIdx.x & 31;
  const int wid  = blockIdx.x * (blockDim.x >> 5) + (threadIdx.x >> 5);
  const int mt = wid >> 6;          // 0..1023 (16-row tiles of B*T = 16384)
  const int ng = wid & 63;          // 0..63   (64-col groups of 4H = 4096)
  const int kh = lane >> 4;
  const int ln = lane & 15;

  const uint16_t* arow = xbf + (size_t)(mt * 16 + ln) * D_ + 8 * kh;
  const uint16_t* brow[4];
#pragma unroll
  for (int s = 0; s < 4; ++s)
    brow[s] = WxT0 + (size_t)(ng * 64 + s * 16 + ln) * D_ + 8 * kh;

  floatx8 acc[4];
#pragma unroll
  for (int s = 0; s < 4; ++s)
#pragma unroll
    for (int r = 0; r < 8; ++r) acc[s][r] = 0.0f;

  // double-buffered K loop (K = 1024, 32 steps, unroll-by-2)
  Frag a0 = ldA(arow, 0), a1;
  Frag b0[4], b1[4];
#pragma unroll
  for (int s = 0; s < 4; ++s) b0[s] = ldB(brow[s], 0);
#pragma unroll 1
  for (int k = 0; k < D_; k += 64) {
    a1 = ldA(arow, k + 32);
#pragma unroll
    for (int s = 0; s < 4; ++s) b1[s] = ldB(brow[s], k + 32);
#pragma unroll
    for (int s = 0; s < 4; ++s)
      acc[s] = __builtin_amdgcn_wmma_f32_16x16x32_bf16(
          false, a0.v, false, b0[s].v, (short)0, acc[s], false, false);
    if (k + 64 < D_) {
      a0 = ldA(arow, k + 64);
#pragma unroll
      for (int s = 0; s < 4; ++s) b0[s] = ldB(brow[s], k + 64);
    }
#pragma unroll
    for (int s = 0; s < 4; ++s)
      acc[s] = __builtin_amdgcn_wmma_f32_16x16x32_bf16(
          false, a1.v, false, b1[s].v, (short)0, acc[s], false, false);
  }

  // store: D layout vgpr r -> row m = r + 8*(lane>=16); col = lane&15
#pragma unroll
  for (int s = 0; s < 4; ++s) {
    const int ncol = ng * 64 + s * 16 + ln;
#pragma unroll
    for (int r = 0; r < 8; ++r) {
      const int j = mt * 16 + r + 8 * kh;     // row in [B*T]
      const int b = j >> 8;                   // j = b*T + t, T = 256
      const int t = j & 255;
      gx0[((size_t)t * B_ + b) * (4 * H_) + ncol] = f32_to_bf16_rne(acc[s][r]);
    }
  }
}

// ---------------- fused LSTM cell ----------------
// Block = 128 threads (4 waves) = one (m,n) 16x16 h-tile with all 4 gates.
// Each wave reduces a K chunk; partials combined in LDS; 128 threads do the
// elementwise cell update.
// HAS_X=true : K = 2048 (waves 0-1: inp@WxT, waves 2-3: hprev@WhT), chunks 512
// HAS_X=false: K = 1024 (all waves on hprev@WhT), chunks 256; gpre adds the
//              precomputed x@Wx contribution.

template <bool HAS_X>
__global__ __launch_bounds__(128)
void lstm_cell_wmma(const uint16_t* __restrict__ inp, int inp_stride,
                    const uint16_t* __restrict__ hprev,
                    const uint16_t* __restrict__ WxT,
                    const uint16_t* __restrict__ WhT,
                    const uint16_t* __restrict__ gpre,   // [B][4H] bf16 (t slice)
                    const float* __restrict__ bias,      // [4H]
                    float* __restrict__ c_state,         // [B][H]
                    uint16_t* __restrict__ hnext,        // [B][H]
                    float* __restrict__ out_f32)         // [B][H] or null
{
  __shared__ float red[4][4][32][8];   // [wave][gate][lane][r] : 16 KB

  const int tid  = threadIdx.x;
  const int lane = tid & 31;
  const int w    = tid >> 5;
  const int task = blockIdx.x;         // 0..255 : 4 m-tiles x 64 n-tiles
  const int m0 = (task >> 6) * 16;
  const int n0 = (task & 63) * 16;
  const int kh = lane >> 4;
  const int ln = lane & 15;
  const int n  = n0 + ln;

  // per-wave K-chunk assignment (wave-uniform)
  const uint16_t* src; int sstride; const uint16_t* W; int kbase; int klen;
  if (HAS_X) {
    klen = 512;
    if (w < 2) { src = inp;   sstride = inp_stride; W = WxT; kbase = w * 512; }
    else       { src = hprev; sstride = H_;         W = WhT; kbase = (w - 2) * 512; }
  } else {
    klen = 256;
    src = hprev; sstride = H_; W = WhT; kbase = w * 256;
  }

  const uint16_t* arow = src + (size_t)(m0 + ln) * sstride + kbase + 8 * kh;
  const uint16_t* brow[4];
#pragma unroll
  for (int g = 0; g < 4; ++g)
    brow[g] = W + (size_t)(g * H_ + n) * D_ + kbase + 8 * kh;

  floatx8 acc[4];
#pragma unroll
  for (int g = 0; g < 4; ++g)
#pragma unroll
    for (int r = 0; r < 8; ++r) acc[g][r] = 0.0f;

  // double-buffered K loop (klen = 256 or 512, unroll-by-2)
  Frag a0 = ldA(arow, 0), a1;
  Frag b0[4], b1[4];
#pragma unroll
  for (int g = 0; g < 4; ++g) b0[g] = ldB(brow[g], 0);
#pragma unroll 1
  for (int k = 0; k < klen; k += 64) {
    a1 = ldA(arow, k + 32);
#pragma unroll
    for (int g = 0; g < 4; ++g) b1[g] = ldB(brow[g], k + 32);
#pragma unroll
    for (int g = 0; g < 4; ++g)
      acc[g] = __builtin_amdgcn_wmma_f32_16x16x32_bf16(
          false, a0.v, false, b0[g].v, (short)0, acc[g], false, false);
    if (k + 64 < klen) {
      a0 = ldA(arow, k + 64);
#pragma unroll
      for (int g = 0; g < 4; ++g) b0[g] = ldB(brow[g], k + 64);
    }
#pragma unroll
    for (int g = 0; g < 4; ++g)
      acc[g] = __builtin_amdgcn_wmma_f32_16x16x32_bf16(
          false, a1.v, false, b1[g].v, (short)0, acc[g], false, false);
  }

  // dump partials to LDS (8 contiguous floats per (w,g,lane) -> 2x ds_store_b128)
#pragma unroll
  for (int g = 0; g < 4; ++g) {
    float4* dst = (float4*)&red[w][g][lane][0];
    dst[0] = make_float4(acc[g][0], acc[g][1], acc[g][2], acc[g][3]);
    dst[1] = make_float4(acc[g][4], acc[g][5], acc[g][6], acc[g][7]);
  }
  __syncthreads();

  // elementwise cell update: 256 cells, 2 per thread; fixed-order (deterministic) sum
#pragma unroll
  for (int cc = 0; cc < 2; ++cc) {
    const int c = tid + cc * 128;        // 0..255
    const int m = c >> 4;                // row in tile
    const int nl = c & 15;               // col in tile
    const int r  = m & 7;
    const int sl = (m >> 3) * 16 + nl;   // source lane in partials
    const int bi = m0 + m;               // batch row
    const int na = n0 + nl;              // absolute column (within H)

    float gv[4];
#pragma unroll
    for (int g = 0; g < 4; ++g) {
      float s = bias[g * H_ + na];
      if (!HAS_X) s += bf16_to_f32(gpre[(size_t)bi * (4 * H_) + g * H_ + na]);
#pragma unroll
      for (int ww = 0; ww < 4; ++ww) s += red[ww][g][sl][r];
      gv[g] = s;
    }
    const float iv = fast_sigmoid(gv[0]);
    const float fv = fast_sigmoid(gv[1]);
    const float g_ = fast_tanh(gv[2]);
    const float ov = fast_sigmoid(gv[3]);

    const size_t idx = (size_t)bi * H_ + na;
    const float cn = fv * c_state[idx] + iv * g_;
    const float hv = ov * fast_tanh(cn);
    c_state[idx] = cn;
    hnext[idx] = f32_to_bf16_rne(hv);
    if (out_f32) out_f32[idx] = hv;
  }
}

// ---------------------------- host-side launcher ----------------------------

extern "C" void kernel_launch(void* const* d_in, const int* in_sizes, int n_in,
                              void* d_out, int out_size, void* d_ws, size_t ws_size,
                              hipStream_t stream) {
  const float* x    = (const float*)d_in[0];   // [B,T,D]
  const float* Wx   = (const float*)d_in[1];   // [L,D,4H]
  const float* Wh   = (const float*)d_in[2];   // [L,H,4H]
  const float* bias = (const float*)d_in[3];   // [L,4H]
  float* out = (float*)d_out;                  // [T,B,H]

  // Workspace carve-up
  uint16_t* xbf    = (uint16_t*)d_ws;                             // B*T*D bf16
  uint16_t* WxT    = xbf  + (size_t)B_ * T_ * D_;                 // L*4H*D bf16
  uint16_t* WhT    = WxT  + (size_t)L_ * 4 * H_ * D_;             // L*4H*H bf16
  uint16_t* hstate = WhT  + (size_t)L_ * 4 * H_ * H_;             // [2][L][B][H] bf16
  float*    cstate = (float*)(hstate + (size_t)2 * L_ * B_ * H_); // [L][B][H] f32
  uint16_t* gx0    = (uint16_t*)(cstate + (size_t)L_ * B_ * H_);  // [T][B][4H] bf16

  const size_t need_base = ((size_t)B_ * T_ * D_ + 2 * (size_t)L_ * 4 * H_ * D_ +
                            (size_t)2 * L_ * B_ * H_) * 2 +
                           (size_t)L_ * B_ * H_ * 4;
  const size_t need_gx = (size_t)T_ * B_ * 4 * H_ * 2;
  const bool use_pre = (ws_size >= need_base + need_gx);  // host-side, deterministic

  // 1) convert x to bf16
  {
    long n = (long)B_ * T_ * D_;
    cvt_f32_to_bf16<<<(unsigned)((n + 255) / 256), 256, 0, stream>>>(x, xbf, n);
  }
  // 2) transpose+convert weights -> [L][4H][K] bf16 (L2-resident thereafter)
  {
    long n = (long)L_ * D_ * 4 * H_;
    cvt_transpose_bf16<<<(unsigned)((n + 255) / 256), 256, 0, stream>>>(Wx, WxT, D_, 4 * H_);
    cvt_transpose_bf16<<<(unsigned)((n + 255) / 256), 256, 0, stream>>>(Wh, WhT, H_, 4 * H_);
  }
  // 3) zero h (both parities) + c — every call (determinism; ws is poisoned once)
  {
    long words = ((size_t)2 * L_ * B_ * H_ * sizeof(uint16_t) +
                  (size_t)L_ * B_ * H_ * sizeof(float)) / 4;
    zero_u32<<<(unsigned)((words + 255) / 256), 256, 0, stream>>>((uint32_t*)hstate, words);
  }
  // 4) hoist layer-0 input GEMM off the critical path (full-occupancy WMMA GEMM)
  if (use_pre) {
    gemm_x_wx0<<<8192, 256, 0, stream>>>(xbf, WxT, gx0);  // 65536 waves
  }

  const size_t layerW  = (size_t)4 * H_ * D_;
  const size_t stateSz = (size_t)L_ * B_ * H_;

  // 5) sequential scan: one fused cell kernel per (t, layer)
  for (int t = 0; t < T_; ++t) {
    const int p  = t & 1;
    const int np = 1 - p;
    for (int l = 0; l < L_; ++l) {
      const uint16_t* hprev = hstate + p  * stateSz + (size_t)l * B_ * H_;
      uint16_t*       hnext = hstate + np * stateSz + (size_t)l * B_ * H_;
      float* outp = (l == L_ - 1) ? (out + (size_t)t * B_ * H_) : (float*)nullptr;
      float* cst  = cstate + (size_t)l * B_ * H_;
      const float* bl = bias + (size_t)l * 4 * H_;

      if (l == 0 && use_pre) {
        lstm_cell_wmma<false><<<256, 128, 0, stream>>>(
            nullptr, 0, hprev, nullptr, WhT, gx0 + (size_t)t * B_ * 4 * H_,
            bl, cst, hnext, outp);
      } else {
        const uint16_t* inp;
        int stride;
        if (l == 0) { inp = xbf + (size_t)t * D_;                          stride = T_ * D_; }
        else        { inp = hstate + np * stateSz + (size_t)(l-1)*B_*H_;   stride = H_;      }
        lstm_cell_wmma<true><<<256, 128, 0, stream>>>(
            inp, stride, hprev, WxT + (size_t)l * layerW, WhT + (size_t)l * layerW,
            nullptr, bl, cst, hnext, outp);
      }
    }
  }
}
